// PostProcessor_81106162418451
// MI455X (gfx1250) — compile-verified, matrix operand order checked by reading, not run on hardware
//
#include <hip/hip_runtime.h>
#include <hip/hip_bf16.h>
#include <math.h>

typedef __attribute__((ext_vector_type(16))) _Float16 v16h;
typedef __attribute__((ext_vector_type(8)))  float    v8f;

#define NROWS 262144
#define NCLS  21
#define NFG   20
#define TOPK  512
#define CAP   2048
#define DET   100
#define SCORE_THRESH 0.05f
#define NMS_THRESH   0.5f
#define BBOX_CLIP    4.135166556742356f   /* log(1000/16) */
#define NEG_BIG      -1000000000.0f

// ---- CDNA5 async global->LDS path (guarded so compile can't regress) ------
#if defined(__has_builtin)
#if __has_builtin(__builtin_amdgcn_global_load_async_to_lds_b128) && \
    __has_builtin(__builtin_amdgcn_s_wait_asynccnt)
#define HAVE_ASYNC_LDS 1
#endif
#endif
#ifndef HAVE_ASYNC_LDS
#define HAVE_ASYNC_LDS 0
#endif

#if HAVE_ASYNC_LDS
typedef __attribute__((vector_size(16))) int v4i_vs;
typedef __attribute__((address_space(1))) v4i_vs g1_v4i;
typedef __attribute__((address_space(3))) v4i_vs l3_v4i;
// copy `nflt` floats (multiple of 4, 16B-aligned src/dst) global -> LDS
__device__ __forceinline__ void async_copy_f32(float* lds_dst,
                                               const float* g_src, int nflt,
                                               int tid, int nthreads) {
    for (int i = tid * 4; i < nflt; i += nthreads * 4) {
        __builtin_amdgcn_global_load_async_to_lds_b128(
            (g1_v4i*)(g_src + i), (l3_v4i*)(lds_dst + i), 0, 0);
    }
    __builtin_amdgcn_s_wait_asynccnt(0);
}
#else
__device__ __forceinline__ void async_copy_f32(float* lds_dst,
                                               const float* g_src, int nflt,
                                               int tid, int nthreads) {
    for (int i = tid; i < nflt; i += nthreads) lds_dst[i] = g_src[i];
}
#endif

// ---------------------------------------------------------------------------
// 0. zero the histogram / counter region of the workspace (replay-safe)
// ---------------------------------------------------------------------------
__global__ void zero_kernel(int* hist, int* cnt) {
    int i = blockIdx.x * blockDim.x + threadIdx.x;
    if (i < NFG * 1024) hist[i] = 0;
    if (i < NFG)        cnt[i]  = 0;
}

// ---------------------------------------------------------------------------
// 1. softmax: one wave handles a 16-row x 32-class (zero padded) tile.
//    Row sums of exp() computed on the matrix unit via
//    v_wmma_f32_16x16x32_f16 with an all-ones B matrix.
//    Writes fg probabilities class-major: scores[c][row], c = class-1.
// ---------------------------------------------------------------------------
__global__ void softmax_kernel(const float* __restrict__ logits,
                               float* __restrict__ scores) {
    const int wave = threadIdx.x >> 5;     // 8 waves / block
    const int lane = threadIdx.x & 31;
    const int m    = lane & 15;
    const int row  = (blockIdx.x * 8 + wave) * 16 + m;
    const int kbase = (lane & 16) ? 8 : 0; // A-fragment K offset (ISA 7.12.2)

    __builtin_prefetch(&logits[(size_t)row * NCLS], 0, 0);

    float v[16];
    float pmax = -INFINITY;
#pragma unroll
    for (int e = 0; e < 16; ++e) {
        int K = e + ((e >= 8) ? 8 : 0) + kbase;
        float x = (K < NCLS) ? logits[(size_t)row * NCLS + K] : -INFINITY;
        v[e] = x;
        pmax = fmaxf(pmax, x);
    }
    float rmax = fmaxf(pmax, __shfl_xor(pmax, 16));

    v16h a;
#pragma unroll
    for (int e = 0; e < 16; ++e) {
        int K = e + ((e >= 8) ? 8 : 0) + kbase;
        float ex = (K < NCLS) ? __expf(v[e] - rmax) : 0.0f;
        _Float16 h = (_Float16)ex;
        a[e] = h;
        v[e] = (float)h;
    }
    v16h b;
#pragma unroll
    for (int e = 0; e < 16; ++e) b[e] = (_Float16)1.0f;

    v8f c = {};
    c = __builtin_amdgcn_wmma_f32_16x16x32_f16(false, a, false, b,
                                               (short)0, c, false, false);

    __shared__ float rowsum[8][16];
    if (lane == 0 || lane == 16) {
        int mb = (lane == 0) ? 0 : 8;
#pragma unroll
        for (int r = 0; r < 8; ++r) rowsum[wave][mb + r] = c[r];
    }
    __syncthreads();

    float inv = 1.0f / rowsum[wave][m];
#pragma unroll
    for (int e = 0; e < 16; ++e) {
        int K = e + ((e >= 8) ? 8 : 0) + kbase;
        if (K >= 1 && K < NCLS)
            scores[(size_t)(K - 1) * NROWS + row] = v[e] * inv;
    }
}

// ---------------------------------------------------------------------------
// 2. per-class 1024-bin histogram of scores
// ---------------------------------------------------------------------------
__global__ void hist_kernel(const float* __restrict__ scores,
                            int* __restrict__ hist) {
    const int c = blockIdx.y;
    __shared__ int h[1024];
    for (int i = threadIdx.x; i < 1024; i += blockDim.x) h[i] = 0;
    __syncthreads();
    for (int i = blockIdx.x * blockDim.x + threadIdx.x; i < NROWS;
         i += gridDim.x * blockDim.x) {
        float s = scores[(size_t)c * NROWS + i];
        int b = (int)(s * 1024.0f);
        b = b < 0 ? 0 : (b > 1023 ? 1023 : b);
        atomicAdd(&h[b], 1);
    }
    __syncthreads();
    for (int i = threadIdx.x; i < 1024; i += blockDim.x)
        if (h[i]) atomicAdd(&hist[c * 1024 + i], h[i]);
}

// ---------------------------------------------------------------------------
// 3. per-class selection threshold (bin index, stored as float)
// ---------------------------------------------------------------------------
__global__ void thresh_kernel(const int* __restrict__ hist,
                              float* __restrict__ thr) {
    const int c = blockIdx.x;
    if (threadIdx.x == 0) {
        int cum = 0, bsel = 0;
        for (int b = 1023; b >= 0; --b) {
            cum += hist[c * 1024 + b];
            if (cum >= TOPK) { bsel = b; break; }
        }
        thr[c] = (float)bsel;
    }
}

// ---------------------------------------------------------------------------
// 4. compact candidates above threshold
// ---------------------------------------------------------------------------
__global__ void collect_kernel(const float* __restrict__ scores,
                               const float* __restrict__ thr,
                               int* __restrict__ cnt,
                               float* __restrict__ candS,
                               int* __restrict__ candI) {
    const int c = blockIdx.y;
    const float t = thr[c];
    for (int i = blockIdx.x * blockDim.x + threadIdx.x; i < NROWS;
         i += gridDim.x * blockDim.x) {
        float s = scores[(size_t)c * NROWS + i];
        if (s > SCORE_THRESH && (s * 1024.0f) >= t) {
            int pos = atomicAdd(&cnt[c], 1);
            if (pos < CAP) {
                candS[c * CAP + pos] = s;
                candI[c * CAP + pos] = i;
            }
        }
    }
}

// ---------------------------------------------------------------------------
// 5. per-class bitonic sort (desc score, asc index) + decode top 512 boxes
//    candS/candI staged into LDS via async global->LDS DMA.
// ---------------------------------------------------------------------------
__global__ void sort_decode_kernel(const float* __restrict__ candS,
                                   const int* __restrict__ candI,
                                   const int* __restrict__ cnt,
                                   const float* __restrict__ boxreg,
                                   const float* __restrict__ props,
                                   float* __restrict__ topkS,
                                   int* __restrict__ validA,
                                   float* __restrict__ candBox) {
    const int c = blockIdx.x;
    const int tid = threadIdx.x;                 // 512 threads = 16 waves
    __shared__ __align__(16) float key[CAP];
    __shared__ __align__(16) int   val[CAP];

    int n = cnt[c]; if (n > CAP) n = CAP;
    // stage raw candidate arrays (stale tail fixed up below)
    async_copy_f32(key, candS + c * CAP, CAP, tid, blockDim.x);
    async_copy_f32((float*)val, (const float*)(candI + c * CAP), CAP, tid,
                   blockDim.x);
    __syncthreads();
    for (int t = n + tid; t < CAP; t += blockDim.x) {
        key[t] = -INFINITY;
        val[t] = 0x7FFFFFFF;
    }

    for (unsigned k = 2; k <= CAP; k <<= 1) {
        for (unsigned j = k >> 1; j > 0; j >>= 1) {
            __syncthreads();
            for (unsigned t = tid; t < CAP; t += blockDim.x) {
                unsigned ixj = t ^ j;
                if (ixj > t) {
                    float sa = key[t], sb = key[ixj];
                    int   ia = val[t], ib = val[ixj];
                    bool a_better = (sa > sb) || (sa == sb && ia < ib);
                    bool desc = ((t & k) == 0);
                    if (desc ? !a_better : a_better) {
                        key[t] = sb; key[ixj] = sa;
                        val[t] = ib; val[ixj] = ia;
                    }
                }
            }
        }
    }
    __syncthreads();

    for (int k2 = tid; k2 < TOPK; k2 += blockDim.x) {
        float s = key[k2];
        int   i = val[k2];
        bool  valid = (s > SCORE_THRESH) && (i != 0x7FFFFFFF);
        int idx = c * TOPK + k2;
        topkS[idx]  = s;
        validA[idx] = valid ? 1 : 0;
        int ii = (i == 0x7FFFFFFF) ? 0 : i;

        const float* p = props + (size_t)ii * 7;
        float px = p[0], py = p[1], pzb = p[2];
        float psx = p[3], psy = p[4], psz = p[5], pry = p[6];
        float pcz = pzb + psz * 0.5f;
        const float* d = boxreg + (size_t)ii * (NCLS * 7) + (c + 1) * 7;
        float dx = d[0] / 10.0f, dy = d[1] / 10.0f, dz = d[2] / 10.0f;
        float dsx = fminf(d[3] / 5.0f, BBOX_CLIP);
        float dsy = fminf(d[4] / 5.0f, BBOX_CLIP);
        float dsz = fminf(d[5] / 5.0f, BBOX_CLIP);
        float cx = px + dx * psx, cy = py + dy * psy, cz = pcz + dz * psz;
        float sx = psx * __expf(dsx), sy = psy * __expf(dsy), sz = psz * __expf(dsz);
        float ry = pry + d[6];
        float zb = cz - sz * 0.5f;
        float* o = candBox + (size_t)idx * 7;
        o[0] = cx; o[1] = cy; o[2] = zb; o[3] = sx; o[4] = sy; o[5] = sz; o[6] = ry;
    }
}

// ---------------------------------------------------------------------------
// 6. per-class NMS: boxes staged via async DMA, IoU bitmasks in LDS,
//    then serial keep scan
// ---------------------------------------------------------------------------
__global__ void nms_kernel(const float* __restrict__ candBox,
                           const float* __restrict__ topkS,
                           const int* __restrict__ validA,
                           float* __restrict__ flatS) {
    const int c = blockIdx.x;
    const int tid = threadIdx.x;                 // 512 threads
    __shared__ __align__(16) float bb[TOPK * 7]; // raw boxes (async staged)
    __shared__ float xl[TOPK], xh[TOPK], yl[TOPK], yh[TOPK];
    __shared__ float zl[TOPK], zh[TOPK], vol[TOPK];
    __shared__ unsigned masks[TOPK][16];
    __shared__ unsigned keepm[16];
    __shared__ int validL[TOPK];

    async_copy_f32(bb, candBox + (size_t)c * TOPK * 7, TOPK * 7, tid,
                   blockDim.x);
    __syncthreads();

    {
        const float* b = bb + tid * 7;
        float cx = b[0], cy = b[1], zb = b[2], sx = b[3], sy = b[4], sz = b[5];
        xl[tid] = cx - sx * 0.5f; xh[tid] = cx + sx * 0.5f;
        yl[tid] = cy - sy * 0.5f; yh[tid] = cy + sy * 0.5f;
        zl[tid] = zb;             zh[tid] = zb + sz;
        vol[tid] = sx * sy * sz;
        validL[tid] = validA[c * TOPK + tid];
    }
    __syncthreads();

    {
        unsigned mw[16];
#pragma unroll
        for (int w = 0; w < 16; ++w) mw[w] = 0u;
        float axl = xl[tid], axh = xh[tid], ayl = yl[tid], ayh = yh[tid];
        float azl = zl[tid], azh = zh[tid], av = vol[tid];
        for (int j = 0; j < TOPK; ++j) {
            float ox = fmaxf(fminf(axh, xh[j]) - fmaxf(axl, xl[j]), 0.0f);
            float oy = fmaxf(fminf(ayh, yh[j]) - fmaxf(ayl, yl[j]), 0.0f);
            float oz = fmaxf(fminf(azh, zh[j]) - fmaxf(azl, zl[j]), 0.0f);
            float inter = ox * oy * oz;
            float iou = inter / (av + vol[j] - inter + 1e-7f);
            if (iou > NMS_THRESH) mw[j >> 5] |= (1u << (j & 31));
        }
#pragma unroll
        for (int w = 0; w < 16; ++w) masks[tid][w] = mw[w];
    }
    __syncthreads();

    if (tid == 0) {
        unsigned km[16];
#pragma unroll
        for (int w = 0; w < 16; ++w) km[w] = 0u;
        for (int i = 0; i < TOPK; ++i) {
            bool sup = false;
            int wlim = i >> 5;
            for (int w = 0; w <= wlim; ++w) {
                unsigned below = (w < wlim) ? 0xFFFFFFFFu
                               : ((i & 31) ? ((1u << (i & 31)) - 1u) : 0u);
                if (masks[i][w] & km[w] & below) { sup = true; break; }
            }
            if (validL[i] && !sup) km[i >> 5] |= (1u << (i & 31));
        }
#pragma unroll
        for (int w = 0; w < 16; ++w) keepm[w] = km[w];
    }
    __syncthreads();

    bool kept = (keepm[tid >> 5] >> (tid & 31)) & 1u;
    flatS[c * TOPK + tid] = kept ? topkS[c * TOPK + tid] : NEG_BIG;
}

// ---------------------------------------------------------------------------
// 7. global top-100 by iterative argmax (tie-break: lowest index, like top_k)
//    flatS staged via async global->LDS DMA.
// ---------------------------------------------------------------------------
__global__ void final_kernel(const float* __restrict__ flatS,
                             const float* __restrict__ candBox,
                             float* __restrict__ out) {
    const int tid = threadIdx.x;                 // 1024 threads
    const int TOT = NFG * TOPK;                  // 10240
    __shared__ __align__(16) float sc[NFG * TOPK];
    __shared__ float bs[1024];
    __shared__ int   bi[1024];

    async_copy_f32(sc, flatS, TOT, tid, blockDim.x);
    __syncthreads();

    for (int d = 0; d < DET; ++d) {
        float bests = -3.0e38f;
        int   besti = TOT;
        for (int i = tid; i < TOT; i += blockDim.x) {
            float s = sc[i];
            if (s > bests || (s == bests && i < besti)) { bests = s; besti = i; }
        }
        bs[tid] = bests; bi[tid] = besti;
        __syncthreads();
        for (int off = 512; off > 0; off >>= 1) {
            if (tid < off) {
                float s2 = bs[tid + off]; int i2 = bi[tid + off];
                if (s2 > bs[tid] || (s2 == bs[tid] && i2 < bi[tid])) {
                    bs[tid] = s2; bi[tid] = i2;
                }
            }
            __syncthreads();
        }
        if (tid == 0) {
            int q = bi[0];
            const float* b = candBox + (size_t)q * 7;
            float* o = out + d * 9;
#pragma unroll
            for (int j = 0; j < 7; ++j) o[j] = b[j];
            o[7] = bs[0];
            o[8] = (float)(q / TOPK + 1);
            sc[q] = -3.0e38f;
        }
        __syncthreads();
    }
}

// ---------------------------------------------------------------------------
extern "C" void kernel_launch(void* const* d_in, const int* in_sizes, int n_in,
                              void* d_out, int out_size, void* d_ws, size_t ws_size,
                              hipStream_t stream) {
    (void)in_sizes; (void)n_in; (void)out_size; (void)ws_size;
    const float* logits = (const float*)d_in[0];   // (N, 21)
    const float* boxreg = (const float*)d_in[1];   // (N, 147)
    const float* props  = (const float*)d_in[2];   // (N, 7)
    float* out = (float*)d_out;                    // (100, 9)

    char* ws = (char*)d_ws;
    size_t off = 0;
    float* scores  = (float*)(ws + off); off += (size_t)NFG * NROWS * 4;   // 20 MB
    int*   hist    = (int*)  (ws + off); off += NFG * 1024 * 4;
    int*   cnt     = (int*)  (ws + off); off += 256;
    float* thr     = (float*)(ws + off); off += 256;
    float* candS   = (float*)(ws + off); off += NFG * CAP * 4;
    int*   candI   = (int*)  (ws + off); off += NFG * CAP * 4;
    float* topkS   = (float*)(ws + off); off += NFG * TOPK * 4;
    int*   validA  = (int*)  (ws + off); off += NFG * TOPK * 4;
    float* candBox = (float*)(ws + off); off += (size_t)NFG * TOPK * 7 * 4;
    float* flatS   = (float*)(ws + off); off += NFG * TOPK * 4;

    zero_kernel<<<(NFG * 1024 + 255) / 256, 256, 0, stream>>>(hist, cnt);
    softmax_kernel<<<NROWS / 128, 256, 0, stream>>>(logits, scores);
    hist_kernel<<<dim3(256, NFG), 256, 0, stream>>>(scores, hist);
    thresh_kernel<<<NFG, 32, 0, stream>>>(hist, thr);
    collect_kernel<<<dim3(256, NFG), 256, 0, stream>>>(scores, thr, cnt, candS, candI);
    sort_decode_kernel<<<NFG, 512, 0, stream>>>(candS, candI, cnt, boxreg, props,
                                                topkS, validA, candBox);
    nms_kernel<<<NFG, 512, 0, stream>>>(candBox, topkS, validA, flatS);
    final_kernel<<<1, 1024, 0, stream>>>(flatS, candBox, out);
}
